// FunctionalInferenceECOAT_45243185496454
// MI455X (gfx1250) — compile-verified, matrix operand order checked by reading, not run on hardware
//
#include <hip/hip_runtime.h>

typedef __attribute__((ext_vector_type(16))) _Float16 v16h;
typedef __attribute__((ext_vector_type(8)))  float    v8f;

#define F_L      1.0f
#define F_SIGMA  0.14f
#define F_RFILM0 50.0f
#define F_VR     1.0f
#define F_DT     0.1f
#define NT       2000
#define F_SCALE  1e9f
#define B_TOT    4096
#define NODES    64

__device__ __forceinline__ float sigmoidf_(float x) {
  // 1/(1+exp(-x)); saturates cleanly to 0/1 for |x| huge (SCALE=1e9)
  return 1.0f / (1.0f + __expf(-x));
}
__device__ __forceinline__ float limitf_(float val, float lim) {
  float d = val - lim;
  return d * sigmoidf_(F_SCALE * d) + lim;
}
__device__ __forceinline__ float tradeofff_(float v1, float v2, float coord, float lim) {
  return (v1 - v2) * sigmoidf_(F_SCALE * (coord - lim)) + v2;
}

// One wave = 16 batch rows. Lanes 0-15 and 16-31 both carry row (lane&15)'s
// recurrence state (replicated halves -> EXEC stays all-1s for WMMA).
__global__ __launch_bounds__(128)
void ecoat_scan_kernel(const float* __restrict__ Cv, const float* __restrict__ Kp,
                       const float* __restrict__ jmin,
                       const float* __restrict__ W1, const float* __restrict__ b1,
                       const float* __restrict__ W2, const float* __restrict__ b2,
                       const float* __restrict__ W3, const float* __restrict__ b3,
                       float* __restrict__ out)
{
  const int lane = threadIdx.x & 31;
  const int wib  = threadIdx.x >> 5;              // wave in block (0..3)
  const int wgl  = blockIdx.x * 4 + wib;          // global wave id
  const int half = lane >> 4;                     // K-half for A/B frags, M-half for C/D
  const int mloc = lane & 15;                     // row within 16-tile / column within N-tile
  const int gm   = wgl * 16 + mloc;               // global batch row

  float* out_thk = out;
  float* out_res = out + (size_t)B_TOT * NT;
  float* out_cur = out + 2ull * B_TOT * NT;

  // --- Preload layer-1 weights into the per-lane A-fragment K slots ---
  // 16-bit A 16x32 layout: VGPR v (f16 pair), lanes<16 hold K base 0, lanes>=16 hold +8;
  // v0..3 -> K 0..7 (+half*8), v4..7 -> K 16..23 (+half*8). Two K-chunks (K0=0,32).
  float aW1[32], aB1[32];
  #pragma unroll
  for (int c = 0; c < 2; ++c) {
    #pragma unroll
    for (int e = 0; e < 16; ++e) {
      int v = e >> 1;
      int k = 32 * c + ((v & 4) ? 16 : 0) + ((v & 3) << 1) + (e & 1) + (half << 3);
      aW1[c * 16 + e] = W1[k];
      aB1[c * 16 + e] = b1[k];
    }
  }

  // --- Pre-convert W2 into 8 constant f16 B-fragments (2 K-chunks x 4 N-tiles) ---
  // B 32x16 layout mirrors A with N = lane&15.
  v16h Bf[2][4];
  #pragma unroll
  for (int c = 0; c < 2; ++c) {
    #pragma unroll
    for (int t = 0; t < 4; ++t) {
      #pragma unroll
      for (int e = 0; e < 16; ++e) {
        int v = e >> 1;
        int k = 32 * c + ((v & 4) ? 16 : 0) + ((v & 3) << 1) + (e & 1) + (half << 3);
        int n = mloc + 16 * t;
        Bf[c][t][e] = (_Float16)W2[k * NODES + n];
      }
    }
  }

  float b2v[4], w3v[4];
  #pragma unroll
  for (int t = 0; t < 4; ++t) {
    b2v[t] = b2[mloc + 16 * t];
    w3v[t] = W3[mloc + 16 * t];
  }
  const float b3s = b3[0];

  // --- Per-row physics constants ---
  const float Cv_r = Cv[gm];
  const float K_r  = Kp[gm];
  const float jm_r = jmin[gm];
  const float beta = F_SIGMA * F_VR / (F_SIGMA * F_RFILM0 + F_L);
  const float Qmin = cbrtf(81.0f / (128.0f * beta)) * powf(K_r, 4.0f / 3.0f);

  float res = F_RFILM0, thk = 0.0f, Q = 0.0f;

  if (half == 0) {  // t = 0 initial column
    size_t o = (size_t)gm * NT;
    out_thk[o] = 0.0f;
    out_res[o] = F_RFILM0;
    out_cur[o] = 1e-3f;
  }

  __shared__ float lds_rho[4 * 16];

  for (int i = 0; i < NT - 1; ++i) {
    float BC   = F_VR * F_DT * (float)(i + 1);
    float jcur = F_SIGMA * BC / (F_SIGMA * res + F_L);
    float Qn   = Q + jcur * F_DT;

    // Layer 1 (rank-1) straight into A fragments, ReLU fused, f32->f16
    v16h A0, A1;
    #pragma unroll
    for (int e = 0; e < 16; ++e) {
      A0[e] = (_Float16)fmaxf(fmaf(jcur, aW1[e],      aB1[e]),      0.0f);
      A1[e] = (_Float16)fmaxf(fmaf(jcur, aW1[16 + e], aB1[16 + e]), 0.0f);
    }

    // Layer 2: h(16x64) @ W2(64x64) via 8 x v_wmma_f32_16x16x32_f16,
    // fused with layer-3 partial dot against W3 in the C/D layout.
    float part[8];
    #pragma unroll
    for (int r = 0; r < 8; ++r) part[r] = 0.0f;

    #pragma unroll
    for (int t = 0; t < 4; ++t) {
      v8f acc = {};
      acc = __builtin_amdgcn_wmma_f32_16x16x32_f16(false, A0, false, Bf[0][t],
                                                   (short)0, acc, false, false);
      acc = __builtin_amdgcn_wmma_f32_16x16x32_f16(false, A1, false, Bf[1][t],
                                                   (short)0, acc, false, false);
      #pragma unroll
      for (int r = 0; r < 8; ++r) {
        float h2 = fmaxf(acc[r] + b2v[t], 0.0f);  // D[r] = row (r + 8*half), col (mloc+16t)
        part[r] = fmaf(h2, w3v[t], part[r]);
      }
    }

    // Reduce over N within each 16-lane half (rows r+8*half)
    #pragma unroll
    for (int mask = 1; mask <= 8; mask <<= 1) {
      #pragma unroll
      for (int r = 0; r < 8; ++r)
        part[r] += __shfl_xor(part[r], mask, 32);
    }

    // Stage rho through LDS to re-align M-split (C/D layout) to lane-owned rows
    if (mloc == 0) {
      #pragma unroll
      for (int r = 0; r < 8; ++r)
        lds_rho[wib * 16 + half * 8 + r] = part[r] + b3s;
    }
    __syncthreads();
    float rho = lds_rho[wib * 16 + mloc];
    __syncthreads();

    // State update (limit + tradeoff), kept in f32 registers
    float dj    = (jcur - jm_r) * F_DT;
    float thk_n = tradeofff_(limitf_(thk + Cv_r * dj, 0.0f),            thk, Qn, Qmin);
    float res_n = tradeofff_(limitf_(res + rho * dj * Cv_r, F_RFILM0),  res, Qn, Qmin);
    res = res_n; thk = thk_n; Q = Qn;

    if (half == 0) {
      size_t o = (size_t)gm * NT + (size_t)(i + 1);
      out_thk[o] = thk_n;
      out_res[o] = res_n;
      out_cur[o] = jcur;
    }
  }
}

__global__ void fill_time_kernel(float* __restrict__ tout) {
  long long idx   = (long long)blockIdx.x * blockDim.x + threadIdx.x;
  long long total = (long long)B_TOT * NT;
  if (idx < total) tout[idx] = F_DT * (float)(idx % NT);
}

extern "C" void kernel_launch(void* const* d_in, const int* in_sizes, int n_in,
                              void* d_out, int out_size, void* d_ws, size_t ws_size,
                              hipStream_t stream) {
  const float* Cv  = (const float*)d_in[0];
  const float* Kp  = (const float*)d_in[1];
  const float* jmn = (const float*)d_in[2];
  const float* W1  = (const float*)d_in[3];
  const float* b1  = (const float*)d_in[4];
  const float* W2  = (const float*)d_in[5];
  const float* b2  = (const float*)d_in[6];
  const float* W3  = (const float*)d_in[7];
  const float* b3  = (const float*)d_in[8];
  float* out = (float*)d_out;

  // 64 rows per block (4 waves x 16 rows) -> 64 blocks cover B=4096
  ecoat_scan_kernel<<<B_TOT / 64, 128, 0, stream>>>(Cv, Kp, jmn, W1, b1, W2, b2,
                                                    W3, b3, out);

  long long total = (long long)B_TOT * NT;
  fill_time_kernel<<<(int)((total + 255) / 256), 256, 0, stream>>>(
      out + 3ull * B_TOT * NT);
}
